// HierarchicalGNNAttention_7559142441637
// MI455X (gfx1250) — compile-verified, matrix operand order checked by reading, not run on hardware
//
#include <hip/hip_runtime.h>
#include <hip/hip_bf16.h>

typedef __attribute__((ext_vector_type(16))) _Float16 v16h;
typedef __attribute__((ext_vector_type(8)))  float    v8f;

#define N_NODES 100000
#define LAT 128
#define HID 256
#define EMB 24
#define M_CLU 1024
#define KB 8
#define KS 8
#define E2 (2 * M_CLU * KS)   // 16384

// ---------------------------------------------------------------------------
// CDNA5 async global->LDS copy (ASYNCcnt path), inline asm per ISA §15.18.3.
// VDST = per-lane LDS byte address, VADDR = per-lane 64-bit global address.
// ---------------------------------------------------------------------------
__device__ __forceinline__ unsigned lds_addr_of(const void* p) {
  // LDS generic addresses carry the wave-relative LDS byte offset in [31:0].
  return (unsigned)(unsigned long long)(uintptr_t)p;
}
__device__ __forceinline__ void async_copy_b128(unsigned ldsOff, const void* gaddr) {
  asm volatile("global_load_async_to_lds_b128 %0, %1, off"
               :: "v"(ldsOff), "v"((unsigned long long)(uintptr_t)gaddr)
               : "memory");
}
__device__ __forceinline__ void async_wait0() {
  asm volatile("s_wait_asynccnt 0" ::: "memory");
}

// ---------------------------------------------------------------------------
// WMMA fragment loader.
// A-matrix 16x32 f16 layout (ISA 7.12.2): lane m = lane&15, half h = lane>>4,
// VGPR q in 0..3 -> K = h*8 + 2q {+0,+1}; q in 4..7 -> K = 16 + h*8 + 2(q-4).
// B (32x16, K-major per column n) mirrors this with n = lane&15 when the
// weight matrix is stored transposed [n][k] (k contiguous) -> dword loads.
// ---------------------------------------------------------------------------
__device__ __forceinline__ v16h load_frag(const _Float16* base, int strideHalves,
                                          int lane, int K0) {
  const int m = lane & 15;
  const int h = lane >> 4;
  union U { v16h v; unsigned int u[8]; } r;
  const _Float16* p = base + (size_t)m * strideHalves + K0 + h * 8;
#pragma unroll
  for (int q = 0; q < 8; ++q) {
    const int kk = ((q < 4) ? 0 : 16) + (q & 3) * 2;
    r.u[q] = *reinterpret_cast<const unsigned int*>(p + kk);
  }
  return r.v;
}

// One GEMM layer: C[16][NOUT] = A[16][K] * Wt^T + bias.  Wt stored [NOUT][K] f16.
template <int K, int NOUT>
__device__ __forceinline__ void mm_layer(const _Float16* A, float (*C)[256],
                                         const _Float16* __restrict__ Wt,
                                         const float* __restrict__ bias, int nReal,
                                         int wave, int lane) {
  constexpr int NT = NOUT / 16;
  for (int tile = wave; tile < NT; tile += 8) {
    v8f acc = {0.f, 0.f, 0.f, 0.f, 0.f, 0.f, 0.f, 0.f};
#pragma unroll
    for (int kc = 0; kc < K / 32; ++kc) {
      v16h a = load_frag(A, 256, lane, kc * 32);
      v16h b = load_frag(Wt + (size_t)tile * 16 * K, K, lane, kc * 32);
      acc = __builtin_amdgcn_wmma_f32_16x16x32_f16(false, a, false, b,
                                                   (short)0, acc, false, false);
    }
    const int h = lane >> 4;
    const int col = tile * 16 + (lane & 15);
    const float bb = (col < nReal) ? bias[col] : 0.f;
#pragma unroll
    for (int r = 0; r < 8; ++r) C[r + h * 8][col] = acc[r] + bb;
  }
}

// LayerNorm + ReLU, result -> f16 activation buffer A.
template <int NOUT>
__device__ __forceinline__ void ln_relu_A(float (*C)[256], _Float16 (*A)[256],
                                          float (*red)[16],
                                          const float* __restrict__ g,
                                          const float* __restrict__ be, int tid) {
  const int row = tid >> 4, j = tid & 15;
  float s = 0.f;
  for (int c = j; c < NOUT; c += 16) s += C[row][c];
  red[row][j] = s;
  __syncthreads();
  float mu = 0.f;
#pragma unroll
  for (int k = 0; k < 16; ++k) mu += red[row][k];
  mu *= (1.0f / NOUT);
  __syncthreads();
  float vs = 0.f;
  for (int c = j; c < NOUT; c += 16) { float d = C[row][c] - mu; vs += d * d; }
  red[row][j] = vs;
  __syncthreads();
  float var = 0.f;
#pragma unroll
  for (int k = 0; k < 16; ++k) var += red[row][k];
  var *= (1.0f / NOUT);
  const float inv = rsqrtf(var + 1e-5f);
  for (int c = j; c < NOUT; c += 16) {
    float y = (C[row][c] - mu) * inv * g[c] + be[c];
    A[row][c] = (_Float16)fmaxf(y, 0.f);
  }
  __syncthreads();
}

// LayerNorm + ReLU, result -> f32 output rows (stride NOUT).
template <int NOUT>
__device__ __forceinline__ void ln_relu_Y(float (*C)[256], float (*red)[16],
                                          const float* __restrict__ g,
                                          const float* __restrict__ be, int tid,
                                          float* __restrict__ Yblk) {
  const int row = tid >> 4, j = tid & 15;
  float s = 0.f;
  for (int c = j; c < NOUT; c += 16) s += C[row][c];
  red[row][j] = s;
  __syncthreads();
  float mu = 0.f;
#pragma unroll
  for (int k = 0; k < 16; ++k) mu += red[row][k];
  mu *= (1.0f / NOUT);
  __syncthreads();
  float vs = 0.f;
  for (int c = j; c < NOUT; c += 16) { float d = C[row][c] - mu; vs += d * d; }
  red[row][j] = vs;
  __syncthreads();
  float var = 0.f;
#pragma unroll
  for (int k = 0; k < 16; ++k) var += red[row][k];
  var *= (1.0f / NOUT);
  const float inv = rsqrtf(var + 1e-5f);
  for (int c = j; c < NOUT; c += 16) {
    float y = (C[row][c] - mu) * inv * g[c] + be[c];
    Yblk[(size_t)row * NOUT + c] = fmaxf(y, 0.f);
  }
}

// ---------------------------------------------------------------------------
// 3-layer MLP: IN -> H (LN+ReLU) -> H (LN+ReLU) -> OUTP (opt LN+ReLU).
// One workgroup (256 threads = 8 waves) per 16 rows.
// Input tile is staged global->LDS with async b128 copies (C as f32 staging),
// then converted to f16 in A.
// ---------------------------------------------------------------------------
template <int IN, int H, int OUTP, int OUTR, bool FACT>
__global__ void __launch_bounds__(256) mlp3_kernel(
    const float* __restrict__ X,
    const _Float16* __restrict__ W1t, const float* __restrict__ b1,
    const float* __restrict__ g1, const float* __restrict__ be1,
    const _Float16* __restrict__ W2t, const float* __restrict__ b2,
    const float* __restrict__ g2, const float* __restrict__ be2,
    const _Float16* __restrict__ W3t, const float* __restrict__ b3,
    const float* __restrict__ g3, const float* __restrict__ be3,
    float* __restrict__ Y) {
  __shared__ __align__(16) _Float16 A[16][256];
  __shared__ __align__(16) float C[16][256];
  __shared__ float red[16][16];
  const int tid = threadIdx.x, lane = tid & 31, wave = tid >> 5;
  const size_t base = (size_t)blockIdx.x * 16;

  // async-stage the 16xIN f32 tile into C (flat), ASYNCcnt tracked
  {
    const char* src = (const char*)(X + base * IN);
    const unsigned cbase = lds_addr_of(&C[0][0]);
    for (int i = tid; i < (16 * IN) / 4; i += 256)
      async_copy_b128(cbase + i * 16, src + (size_t)i * 16);
    async_wait0();
  }
  __syncthreads();
  {
    const float* Cf = &C[0][0];
    for (int i = tid; i < 16 * IN; i += 256) {
      const int r = i / IN, c = i % IN;
      A[r][c] = (_Float16)Cf[i];
    }
  }
  __syncthreads();

  mm_layer<IN, H>(&A[0][0], C, W1t, b1, H, wave, lane);
  __syncthreads();
  ln_relu_A<H>(C, A, red, g1, be1, tid);

  mm_layer<H, H>(&A[0][0], C, W2t, b2, H, wave, lane);
  __syncthreads();
  ln_relu_A<H>(C, A, red, g2, be2, tid);

  mm_layer<H, OUTP>(&A[0][0], C, W3t, b3, OUTR, wave, lane);
  __syncthreads();
  if constexpr (FACT) {
    ln_relu_Y<OUTR>(C, red, g3, be3, tid, Y + base * OUTR);
  } else {
    for (int i = tid; i < 16 * OUTR; i += 256) {
      const int r = i / OUTR, c = i % OUTR;
      Y[(base + r) * OUTR + c] = C[r][c];
    }
  }
}

// ---------------------------------------------------------------------------
// Weight convert + transpose:  W[k][n] f32 -> Wt[n][k] f16 (n padded).
// ---------------------------------------------------------------------------
__global__ void wconv_kernel(const float* __restrict__ W, _Float16* __restrict__ Wt,
                             int K, int Nout, int NoutPad) {
  const long i = (long)blockIdx.x * blockDim.x + threadIdx.x;
  if (i >= (long)K * NoutPad) return;
  const int n = (int)(i / K), k = (int)(i % K);
  Wt[(size_t)n * K + k] = (n < Nout) ? (_Float16)W[(size_t)k * Nout + n] : (_Float16)0.f;
}

__global__ void init_kernel(float* o_sup, float* msum, float* cnt, float* bn,
                            float* mx, float* mn) {
  const int i = blockIdx.x * blockDim.x + threadIdx.x;
  if (i < M_CLU * LAT) o_sup[i] = 0.f;
  if (i < M_CLU * EMB) msum[i] = 0.f;
  if (i < M_CLU) { cnt[i] = 0.f; mx[i] = -1e30f; mn[i] = 1e30f; }
  if (i < 4) bn[i] = 0.f;
}

__global__ void l2norm_emb_kernel(float* __restrict__ emb, _Float16* __restrict__ emb16) {
  const long n = (long)blockIdx.x * blockDim.x + threadIdx.x;
  if (n >= N_NODES) return;
  float s = 0.f;
  for (int d = 0; d < EMB; ++d) { float v = emb[n * EMB + d]; s += v * v; }
  const float inv = 1.f / fmaxf(sqrtf(s), 1e-12f);
  for (int d = 0; d < EMB; ++d) {
    const float v = emb[n * EMB + d] * inv;
    emb[n * EMB + d] = v;
    emb16[n * 32 + d] = (_Float16)v;
  }
  for (int d = EMB; d < 32; ++d) emb16[n * 32 + d] = (_Float16)0.f;
}

__global__ void scatter_mean_kernel(const float* __restrict__ emb,
                                    const int* __restrict__ clusters,
                                    float* __restrict__ msum, float* __restrict__ cnt) {
  const long n = (long)blockIdx.x * blockDim.x + threadIdx.x;
  if (n >= N_NODES) return;
  const int cl = clusters[n];
  for (int d = 0; d < EMB; ++d) atomicAdd(&msum[(size_t)cl * EMB + d], emb[n * EMB + d]);
  atomicAdd(&cnt[cl], 1.f);
}

__global__ void means_fin_kernel(const float* __restrict__ msum,
                                 const float* __restrict__ cnt,
                                 float* __restrict__ meansf,
                                 _Float16* __restrict__ means16) {
  const int m = blockIdx.x * blockDim.x + threadIdx.x;
  if (m >= M_CLU) return;
  const float ic = 1.f / fmaxf(cnt[m], 1.f);
  float v[EMB], s = 0.f;
  for (int d = 0; d < EMB; ++d) { v[d] = msum[m * EMB + d] * ic; s += v[d] * v[d]; }
  const float inv = 1.f / fmaxf(sqrtf(s), 1e-12f);
  for (int d = 0; d < EMB; ++d) {
    const float x = v[d] * inv;
    meansf[m * EMB + d] = x;
    means16[m * 32 + d] = (_Float16)x;
  }
  for (int d = EMB; d < 32; ++d) means16[m * 32 + d] = (_Float16)0.f;
}

// ---------------------------------------------------------------------------
// Shared top-8 machinery: 16 threads per row scan 64-col strips, then one
// thread per row merges the 128 candidates (ties keep the lowest column).
// ---------------------------------------------------------------------------
__device__ __forceinline__ void topk8_insert(float v, int c, float* bv, int* bi) {
  if (v > bv[KB - 1]) {
    int j = KB - 1;
    while (j > 0 && v > bv[j - 1]) { bv[j] = bv[j - 1]; bi[j] = bi[j - 1]; --j; }
    bv[j] = v; bi[j] = c;
  }
}

// sim = emb @ means^T (WMMA, K=32 padded) + per-row top-8 + BN stage-1 sums.
__global__ void __launch_bounds__(256) sim_topk_kernel(
    const _Float16* __restrict__ emb16, const _Float16* __restrict__ means16,
    const float* __restrict__ embf, const float* __restrict__ meansf,
    float* __restrict__ attRaw, int* __restrict__ idxI,
    float* __restrict__ o_idx, float* __restrict__ o_bmask, float* __restrict__ bn) {
  __shared__ __align__(16) _Float16 S[16][1024];
  __shared__ __align__(16) _Float16 At[16][32];
  __shared__ float candV[16][16 * KB];
  __shared__ int   candI[16][16 * KB];
  const int tid = threadIdx.x, lane = tid & 31, wave = tid >> 5;
  const size_t base = (size_t)blockIdx.x * 16;

  // async-stage the f16 A tile (16x32 halves = 1KB)
  {
    const char* src = (const char*)(emb16 + base * 32);
    const unsigned abase = lds_addr_of(&At[0][0]);
    for (int i = tid; i < 64; i += 256) async_copy_b128(abase + i * 16, src + i * 16);
    async_wait0();
  }
  __syncthreads();

  for (int tile = wave; tile < 64; tile += 8) {
    v16h a = load_frag(&At[0][0], 32, lane, 0);
    v16h b = load_frag(means16 + (size_t)tile * 16 * 32, 32, lane, 0);
    v8f acc = {0.f, 0.f, 0.f, 0.f, 0.f, 0.f, 0.f, 0.f};
    acc = __builtin_amdgcn_wmma_f32_16x16x32_f16(false, a, false, b, (short)0,
                                                 acc, false, false);
    const int h = lane >> 4, col = tile * 16 + (lane & 15);
#pragma unroll
    for (int r = 0; r < 8; ++r) S[r + h * 8][col] = (_Float16)acc[r];
  }
  __syncthreads();

  // parallel partial top-8: 16 threads per row, 64 columns each
  {
    const int row = tid >> 4, seg = tid & 15;
    float bv[KB]; int bi[KB];
#pragma unroll
    for (int k = 0; k < KB; ++k) { bv[k] = -1e30f; bi[k] = 0; }
    for (int c = seg * 64; c < seg * 64 + 64; ++c)
      topk8_insert((float)S[row][c], c, bv, bi);
    for (int k = 0; k < KB; ++k) {
      candV[row][seg * KB + k] = bv[k];
      candI[row][seg * KB + k] = bi[k];
    }
  }
  __syncthreads();

  if ((tid & 15) == 0) {
    const int row = tid >> 4;
    const size_t n = base + row;
    float bv[KB]; int bi[KB];
#pragma unroll
    for (int k = 0; k < KB; ++k) { bv[k] = -1e30f; bi[k] = 0; }
    for (int c = 0; c < 16 * KB; ++c) topk8_insert(candV[row][c], candI[row][c], bv, bi);
    float lsum = 0.f, lcnt = 0.f;
    for (int k = 0; k < KB; ++k) {
      const int c = bi[k];
      float att = 0.f;
      for (int d = 0; d < EMB; ++d) att += embf[n * EMB + d] * meansf[(size_t)c * EMB + d];
      const float bm = (att >= 0.5f) ? 1.f : 0.f;
      attRaw[n * KB + k] = att;
      idxI[n * KB + k] = c;
      o_idx[n * KB + k] = (float)c;
      o_bmask[n * KB + k] = bm;
      lsum += att * bm;
      lcnt += bm;
    }
    atomicAdd(&bn[0], lsum);
    atomicAdd(&bn[1], lcnt);
  }
}

__global__ void bn_var_kernel(const float* __restrict__ attRaw,
                              const float* __restrict__ bmask, float* __restrict__ bn) {
  const long n = (long)blockIdx.x * blockDim.x + threadIdx.x;
  if (n >= N_NODES) return;
  const float c = fmaxf(bn[1], 1.f);
  const float mu = bn[0] / c;
  float s = 0.f;
  for (int k = 0; k < KB; ++k) {
    const float d = (attRaw[n * KB + k] - mu) * bmask[n * KB + k];
    s += d * d;
  }
  atomicAdd(&bn[2], s);
}

__global__ void softmax_kernel(const float* __restrict__ attRaw,
                               const float* __restrict__ bmask,
                               const float* __restrict__ bn,
                               const float* __restrict__ gamma_p,
                               const float* __restrict__ beta_p,
                               float* __restrict__ o_logits,
                               float* __restrict__ o_attw, float* __restrict__ attW) {
  const long n = (long)blockIdx.x * blockDim.x + threadIdx.x;
  if (n >= N_NODES) return;
  const float c = fmaxf(bn[1], 1.f);
  const float mu = bn[0] / c;
  const float var = bn[2] / c;
  const float inv = rsqrtf(var + 1e-5f);
  const float gamma = gamma_p[0], beta = beta_p[0];
  float w[KB], s = 0.f;
  for (int k = 0; k < KB; ++k) {
    const float lg = (attRaw[n * KB + k] - mu) * inv * gamma + beta;
    const float bm = bmask[n * KB + k];
    o_logits[n * KB + k] = lg * bm;
    const float ww = expf(lg) * bm;
    w[k] = ww;
    s += ww;
  }
  const float d = 1.f / (1e-12f + s);
  for (int k = 0; k < KB; ++k) {
    const float aw = w[k] * d;
    o_attw[n * KB + k] = aw;
    attW[n * KB + k] = aw;
  }
}

__global__ void scatter_super_kernel(const float* __restrict__ node_msg,
                                     const float* __restrict__ attW,
                                     const int* __restrict__ idxI,
                                     float* __restrict__ o_sup) {
  const long e = (long)blockIdx.x * blockDim.x + threadIdx.x;
  if (e >= (long)N_NODES * KB) return;
  const float w = attW[e];
  if (w == 0.f) return;
  const long n = e >> 3;
  const int m = idxI[e];
  for (int c = 0; c < LAT; ++c)
    atomicAdd(&o_sup[(size_t)m * LAT + c], node_msg[n * LAT + c] * w);
}

// simm = means @ means^T (WMMA) + top-8 -> src/dst/emask
__global__ void __launch_bounds__(256) simm_topk_kernel(
    const _Float16* __restrict__ means16, const float* __restrict__ meansf,
    int* __restrict__ srcI, int* __restrict__ dstI, float* __restrict__ o_src,
    float* __restrict__ o_dst, float* __restrict__ o_emask) {
  __shared__ __align__(16) _Float16 S[16][1024];
  __shared__ __align__(16) _Float16 At[16][32];
  __shared__ float candV[16][16 * KS];
  __shared__ int   candI[16][16 * KS];
  const int tid = threadIdx.x, lane = tid & 31, wave = tid >> 5;
  const size_t base = (size_t)blockIdx.x * 16;

  {
    const char* src = (const char*)(means16 + base * 32);
    const unsigned abase = lds_addr_of(&At[0][0]);
    for (int i = tid; i < 64; i += 256) async_copy_b128(abase + i * 16, src + i * 16);
    async_wait0();
  }
  __syncthreads();

  for (int tile = wave; tile < 64; tile += 8) {
    v16h a = load_frag(&At[0][0], 32, lane, 0);
    v16h b = load_frag(means16 + (size_t)tile * 16 * 32, 32, lane, 0);
    v8f acc = {0.f, 0.f, 0.f, 0.f, 0.f, 0.f, 0.f, 0.f};
    acc = __builtin_amdgcn_wmma_f32_16x16x32_f16(false, a, false, b, (short)0,
                                                 acc, false, false);
    const int h = lane >> 4, col = tile * 16 + (lane & 15);
#pragma unroll
    for (int r = 0; r < 8; ++r) S[r + h * 8][col] = (_Float16)acc[r];
  }
  __syncthreads();

  {
    const int row = tid >> 4, seg = tid & 15;
    float bv[KS]; int bi[KS];
#pragma unroll
    for (int k = 0; k < KS; ++k) { bv[k] = -1e30f; bi[k] = 0; }
    for (int c = seg * 64; c < seg * 64 + 64; ++c)
      topk8_insert((float)S[row][c], c, bv, bi);
    for (int k = 0; k < KS; ++k) {
      candV[row][seg * KS + k] = bv[k];
      candI[row][seg * KS + k] = bi[k];
    }
  }
  __syncthreads();

  if ((tid & 15) == 0) {
    const int row = tid >> 4;
    const int r0 = (int)base + row;
    float bv[KS]; int bi[KS];
#pragma unroll
    for (int k = 0; k < KS; ++k) { bv[k] = -1e30f; bi[k] = 0; }
    for (int c = 0; c < 16 * KS; ++c) topk8_insert(candV[row][c], candI[row][c], bv, bi);
    for (int k = 0; k < KS; ++k) {
      const int c = bi[k];
      float v = 0.f;
      for (int d = 0; d < EMB; ++d)
        v += meansf[(size_t)r0 * EMB + d] * meansf[(size_t)c * EMB + d];
      const float em = (v >= 0.5f) ? 1.f : 0.f;
      const int e = r0 * KS + k;
      srcI[e] = r0; dstI[e] = c;
      srcI[M_CLU * KS + e] = c; dstI[M_CLU * KS + e] = r0;
      o_src[e] = (float)r0; o_dst[e] = (float)c;
      o_src[M_CLU * KS + e] = (float)c; o_dst[M_CLU * KS + e] = (float)r0;
      o_emask[e] = em; o_emask[M_CLU * KS + e] = em;
    }
  }
}

__device__ __forceinline__ void atomicMaxF(float* addr, float val) {
  unsigned int* ua = reinterpret_cast<unsigned int*>(addr);
  unsigned int old = *ua, assumed;
  do {
    const float f = __uint_as_float(old);
    if (f >= val) return;
    assumed = old;
    old = atomicCAS(ua, assumed, __float_as_uint(val));
  } while (old != assumed);
}
__device__ __forceinline__ void atomicMinF(float* addr, float val) {
  unsigned int* ua = reinterpret_cast<unsigned int*>(addr);
  unsigned int old = *ua, assumed;
  do {
    const float f = __uint_as_float(old);
    if (f <= val) return;
    assumed = old;
    old = atomicCAS(ua, assumed, __float_as_uint(val));
  } while (old != assumed);
}

__global__ void atts_minmax_kernel(const float* __restrict__ meansf,
                                   const int* __restrict__ srcI,
                                   const int* __restrict__ dstI,
                                   const float* __restrict__ o_emask,
                                   float* __restrict__ atts, float* __restrict__ mx,
                                   float* __restrict__ mn) {
  const int e = blockIdx.x * blockDim.x + threadIdx.x;
  if (e >= E2) return;
  const int s = srcI[e], d = dstI[e];
  float a = 0.f;
  for (int k = 0; k < EMB; ++k)
    a += meansf[(size_t)s * EMB + k] * meansf[(size_t)d * EMB + k];
  atts[e] = a;
  const float em = o_emask[e];
  atomicMaxF(&mx[d], (em != 0.f) ? a : -1e30f);
  atomicMinF(&mn[d], (em != 0.f) ? a : 1e30f);
}

__global__ void sup_att_kernel(const float* __restrict__ atts,
                               const int* __restrict__ dstI,
                               const float* __restrict__ o_emask,
                               const float* __restrict__ mx,
                               const float* __restrict__ mn,
                               float* __restrict__ o_supatt) {
  const int e = blockIdx.x * blockDim.x + threadIdx.x;
  if (e >= E2) return;
  const int d = dstI[e];
  const float v = tanhf(2.f * (atts[e] - mn[d]) / (1e-12f + (mx[d] - mn[d])));
  o_supatt[e] = v * o_emask[e];
}

__global__ void ein_gather_kernel(const float* __restrict__ o_sup,
                                  const int* __restrict__ srcI,
                                  const int* __restrict__ dstI,
                                  float* __restrict__ ein) {
  const long i = (long)blockIdx.x * blockDim.x + threadIdx.x;
  if (i >= (long)E2 * 2 * LAT) return;
  const int e = (int)(i >> 8);
  const int c = (int)(i & 255);
  ein[i] = (c < LAT) ? o_sup[(size_t)srcI[e] * LAT + c]
                     : o_sup[(size_t)dstI[e] * LAT + (c - LAT)];
}

__global__ void mask_sedge_kernel(float* __restrict__ o_sedge,
                                  const float* __restrict__ o_emask) {
  const long i = (long)blockIdx.x * blockDim.x + threadIdx.x;
  if (i >= (long)E2 * LAT) return;
  o_sedge[i] *= o_emask[i >> 7];
}

// ---------------------------------------------------------------------------
extern "C" void kernel_launch(void* const* d_in, const int* in_sizes, int n_in,
                              void* d_out, int out_size, void* d_ws, size_t ws_size,
                              hipStream_t stream) {
  (void)in_sizes; (void)n_in; (void)out_size; (void)ws_size;
  const float* embedding_nodes = (const float*)d_in[0];
  const float* encoded_nodes   = (const float*)d_in[1];
  const int*   clusters        = (const int*)d_in[2];
  const float* cluW[3]  = {(const float*)d_in[3],  (const float*)d_in[4],  (const float*)d_in[5]};
  const float* cluB[3]  = {(const float*)d_in[6],  (const float*)d_in[7],  (const float*)d_in[8]};
  const float* cluG[2]  = {(const float*)d_in[9],  (const float*)d_in[10]};
  const float* cluBe[2] = {(const float*)d_in[11], (const float*)d_in[12]};
  const float* nodW[3]  = {(const float*)d_in[13], (const float*)d_in[14], (const float*)d_in[15]};
  const float* nodB[3]  = {(const float*)d_in[16], (const float*)d_in[17], (const float*)d_in[18]};
  const float* nodG[3]  = {(const float*)d_in[19], (const float*)d_in[20], (const float*)d_in[21]};
  const float* nodBe[3] = {(const float*)d_in[22], (const float*)d_in[23], (const float*)d_in[24]};
  const float* edgW[3]  = {(const float*)d_in[25], (const float*)d_in[26], (const float*)d_in[27]};
  const float* edgB[3]  = {(const float*)d_in[28], (const float*)d_in[29], (const float*)d_in[30]};
  const float* edgG[3]  = {(const float*)d_in[31], (const float*)d_in[32], (const float*)d_in[33]};
  const float* edgBe[3] = {(const float*)d_in[34], (const float*)d_in[35], (const float*)d_in[36]};
  const float* bn_gamma = (const float*)d_in[37];
  const float* bn_beta  = (const float*)d_in[38];

  // output layout (all f32, flat in return order)
  float* o_emb    = (float*)d_out;                       // N*EMB
  float* o_sup    = o_emb    + (size_t)N_NODES * EMB;    // M*LAT
  float* o_sedge  = o_sup    + (size_t)M_CLU * LAT;      // E2*LAT
  float* o_logits = o_sedge  + (size_t)E2 * LAT;         // N*KB
  float* o_attw   = o_logits + (size_t)N_NODES * KB;     // N*KB
  float* o_supatt = o_attw   + (size_t)N_NODES * KB;     // E2
  float* o_idx    = o_supatt + (size_t)E2;               // N*KB
  float* o_bmask  = o_idx    + (size_t)N_NODES * KB;     // N*KB
  float* o_src    = o_bmask  + (size_t)N_NODES * KB;     // E2
  float* o_dst    = o_src    + (size_t)E2;               // E2
  float* o_emask  = o_dst    + (size_t)E2;               // E2

  // workspace carve-out
  char* wsp = (char*)d_ws;
  size_t cur = 0;
  auto alloc = [&](size_t bytes) -> void* {
    void* p = wsp + cur;
    cur += (bytes + 255) & ~(size_t)255;
    return p;
  };
  _Float16* cluW1t = (_Float16*)alloc((size_t)256 * 128 * 2);
  _Float16* cluW2t = (_Float16*)alloc((size_t)256 * 256 * 2);
  _Float16* cluW3t = (_Float16*)alloc((size_t)32 * 256 * 2);
  _Float16* nodW1t = (_Float16*)alloc((size_t)256 * 128 * 2);
  _Float16* nodW2t = (_Float16*)alloc((size_t)256 * 256 * 2);
  _Float16* nodW3t = (_Float16*)alloc((size_t)128 * 256 * 2);
  _Float16* edgW1t = (_Float16*)alloc((size_t)256 * 256 * 2);
  _Float16* edgW2t = (_Float16*)alloc((size_t)256 * 256 * 2);
  _Float16* edgW3t = (_Float16*)alloc((size_t)128 * 256 * 2);
  _Float16* emb16   = (_Float16*)alloc((size_t)N_NODES * 32 * 2);
  float*    node_msg = (float*)alloc((size_t)N_NODES * LAT * 4);
  float*    msum     = (float*)alloc((size_t)M_CLU * EMB * 4);
  float*    cntb     = (float*)alloc((size_t)M_CLU * 4);
  float*    meansf   = (float*)alloc((size_t)M_CLU * EMB * 4);
  _Float16* means16  = (_Float16*)alloc((size_t)M_CLU * 32 * 2);
  float*    attRaw   = (float*)alloc((size_t)N_NODES * KB * 4);
  float*    attW     = (float*)alloc((size_t)N_NODES * KB * 4);
  int*      idxI     = (int*)alloc((size_t)N_NODES * KB * 4);
  float*    bnS      = (float*)alloc(4 * 4);
  int*      srcI     = (int*)alloc((size_t)E2 * 4);
  int*      dstI     = (int*)alloc((size_t)E2 * 4);
  float*    attsE    = (float*)alloc((size_t)E2 * 4);
  float*    mxA      = (float*)alloc((size_t)M_CLU * 4);
  float*    mnA      = (float*)alloc((size_t)M_CLU * 4);
  float*    ein      = (float*)alloc((size_t)E2 * 2 * LAT * 4);

  auto blk = [](long n) { return (unsigned)((n + 255) / 256); };

  // 1) f16 weight conversion (transposed, padded)
  wconv_kernel<<<blk(128L * 256), 256, 0, stream>>>(cluW[0], cluW1t, 128, 256, 256);
  wconv_kernel<<<blk(256L * 256), 256, 0, stream>>>(cluW[1], cluW2t, 256, 256, 256);
  wconv_kernel<<<blk(256L * 32), 256, 0, stream>>>(cluW[2], cluW3t, 256, 24, 32);
  wconv_kernel<<<blk(128L * 256), 256, 0, stream>>>(nodW[0], nodW1t, 128, 256, 256);
  wconv_kernel<<<blk(256L * 256), 256, 0, stream>>>(nodW[1], nodW2t, 256, 256, 256);
  wconv_kernel<<<blk(256L * 128), 256, 0, stream>>>(nodW[2], nodW3t, 256, 128, 128);
  wconv_kernel<<<blk(256L * 256), 256, 0, stream>>>(edgW[0], edgW1t, 256, 256, 256);
  wconv_kernel<<<blk(256L * 256), 256, 0, stream>>>(edgW[1], edgW2t, 256, 256, 256);
  wconv_kernel<<<blk(256L * 128), 256, 0, stream>>>(edgW[2], edgW3t, 256, 128, 128);

  // 2) init accumulators
  init_kernel<<<blk(M_CLU * LAT), 256, 0, stream>>>(o_sup, msum, cntb, bnS, mxA, mnA);

  // 3) clu MLP -> raw emb (in d_out), then l2norm (+f16 padded copy)
  mlp3_kernel<128, 256, 32, 24, false><<<N_NODES / 16, 256, 0, stream>>>(
      embedding_nodes, cluW1t, cluB[0], cluG[0], cluBe[0], cluW2t, cluB[1],
      cluG[1], cluBe[1], cluW3t, cluB[2], nullptr, nullptr, o_emb);
  l2norm_emb_kernel<<<blk(N_NODES), 256, 0, stream>>>(o_emb, emb16);

  // 4) node MLP -> node_msg
  mlp3_kernel<128, 256, 128, 128, true><<<N_NODES / 16, 256, 0, stream>>>(
      encoded_nodes, nodW1t, nodB[0], nodG[0], nodBe[0], nodW2t, nodB[1],
      nodG[1], nodBe[1], nodW3t, nodB[2], nodG[2], nodBe[2], node_msg);

  // 5) cluster means
  scatter_mean_kernel<<<blk(N_NODES), 256, 0, stream>>>(o_emb, clusters, msum, cntb);
  means_fin_kernel<<<blk(M_CLU), 256, 0, stream>>>(msum, cntb, meansf, means16);

  // 6) bipartite kNN (WMMA sim + top-8) + BN pass 1
  sim_topk_kernel<<<N_NODES / 16, 256, 0, stream>>>(emb16, means16, o_emb, meansf,
                                                    attRaw, idxI, o_idx, o_bmask, bnS);
  bn_var_kernel<<<blk(N_NODES), 256, 0, stream>>>(attRaw, o_bmask, bnS);
  softmax_kernel<<<blk(N_NODES), 256, 0, stream>>>(attRaw, o_bmask, bnS, bn_gamma,
                                                   bn_beta, o_logits, o_attw, attW);

  // 7) supernodes = segment_sum of weighted messages
  scatter_super_kernel<<<blk((long)N_NODES * KB), 256, 0, stream>>>(node_msg, attW,
                                                                    idxI, o_sup);

  // 8) supergraph kNN + edges
  simm_topk_kernel<<<M_CLU / 16, 256, 0, stream>>>(means16, meansf, srcI, dstI,
                                                   o_src, o_dst, o_emask);
  atts_minmax_kernel<<<blk(E2), 256, 0, stream>>>(meansf, srcI, dstI, o_emask,
                                                  attsE, mxA, mnA);
  sup_att_kernel<<<blk(E2), 256, 0, stream>>>(attsE, dstI, o_emask, mxA, mnA,
                                              o_supatt);

  // 9) edge MLP on concat(supernodes[src], supernodes[dst]) then mask
  ein_gather_kernel<<<blk((long)E2 * 2 * LAT), 256, 0, stream>>>(o_sup, srcI, dstI, ein);
  mlp3_kernel<256, 256, 128, 128, true><<<E2 / 16, 256, 0, stream>>>(
      ein, edgW1t, edgB[0], edgG[0], edgBe[0], edgW2t, edgB[1], edgG[1], edgBe[1],
      edgW3t, edgB[2], edgG[2], edgBe[2], o_sedge);
  mask_sedge_kernel<<<blk((long)E2 * LAT), 256, 0, stream>>>(o_sedge, o_emask);
}
